// QMamba_24524263260373
// MI455X (gfx1250) — 8-bit matrix operand order repaired and checked by reading, compiled, not run on hardware
//
#include <hip/hip_runtime.h>
#include <hip/hip_bf16.h>


typedef signed char i8;
typedef int v8i __attribute__((ext_vector_type(8)));

#define BATCH    2
#define SEQLEN   1024
#define D_MODEL  1024
#define D_INNER  2048
#define D_STATE  16
#define D_CONV   4
#define DT_RANK  64
#define NTOK     (BATCH * SEQLEN)
#define XDIM     (DT_RANK + 2 * D_STATE)

static constexpr float S_IN       = 0.05f;
static constexpr float S_XZ       = 0.06f;
static constexpr float S_CONV_OUT = 0.04f;
static constexpr float S_XDBL     = 0.05f;
static constexpr float S_DT       = 0.05f;
static constexpr float S_Y        = 0.10f;

__device__ __forceinline__ v8i load_a_frag(const i8* __restrict__ A, int lda,
                                           int m0, int kk, int lane) {
    v8i a;
    const int mlane = lane & 15;
    const int hi    = (lane >> 4) & 1;
    const i8* row = A + (size_t)(m0 + mlane) * (size_t)lda + kk;
#pragma unroll
    for (int i = 0; i < 8; ++i) {
        const int kbase = ((i >> 1) << 4) + ((i & 1) << 2) + (hi << 3);
        a[i] = *(const int*)(row + kbase);
    }
    return a;
}

__device__ __forceinline__ v8i load_b_frag(const i8* __restrict__ Bt, int ldb,
                                           int n0, int kk, int lane) {
    v8i b;
    const int col = lane & 15;
    const int hi  = (lane >> 4) & 1;
    const i8* row = Bt + (size_t)(n0 + col) * (size_t)ldb + kk;
#pragma unroll
    for (int i = 0; i < 8; ++i) {
        const int kbase = ((i >> 1) << 4) + ((i & 1) << 2) + (hi << 3);
        b[i] = *(const int*)(row + kbase);
    }
    return b;
}

__device__ __forceinline__ float softplusf(float x) {
    return (x > 20.0f) ? x : log1pf(__expf(x));
}
__device__ __forceinline__ float siluf(float x) {
    return x / (1.0f + __expf(-x));
}
__device__ __forceinline__ float fq_clampf(float v, float s) {
    return fminf(fmaxf(rintf(v / s), -128.0f), 127.0f);
}

template <int MODE, int NT>
__global__ void gemm_i8_wmma(const i8* __restrict__ A, int lda,
                             const i8* __restrict__ Bt, int ldb,
                             int M, int N, int K,
                             float sA, const unsigned* __restrict__ wmax,
                             float outScale,
                             const float* __restrict__ bias,
                             i8* __restrict__ out8,
                             float* __restrict__ outf, int ldo) {
    const int lane   = threadIdx.x & 31;
    const int wave   = threadIdx.x >> 5;
    const int group  = blockIdx.x * (blockDim.x >> 5) + wave;
    const int ngrp_n = N / (16 * NT);
    const int total  = (M >> 4) * ngrp_n;
    if (group >= total) return;
    const int tm = group / ngrp_n;
    const int n0 = (group % ngrp_n) * (16 * NT);
    const int m0 = tm << 4;

    const v8i zero = {0, 0, 0, 0, 0, 0, 0, 0};
    v8i acc[NT];
#pragma unroll
    for (int t = 0; t < NT; ++t) acc[t] = zero;

    for (int kk = 0; kk < K; kk += 64) {
        if (kk + 64 < K) {
            __builtin_prefetch(A + (size_t)(m0 + (lane & 15)) * lda + kk + 64, 0, 1);
            __builtin_prefetch(Bt + (size_t)(n0 + lane) * ldb + kk + 64, 0, 1);
        }
        const v8i a = load_a_frag(A, lda, m0, kk, lane);
#pragma unroll
        for (int t = 0; t < NT; ++t) {
            const v8i b = load_b_frag(Bt, ldb, n0 + 16 * t, kk, lane);
            acc[t] = __builtin_amdgcn_wmma_i32_16x16x64_iu8(
                 true, a,  true, b, acc[t],
                 false,  false);
        }
    }

    const float sB  = __uint_as_float(*wmax) * (1.0f / 127.0f);
    const float deq = sA * sB;
    const int hi  = lane >> 4;
    const int col = lane & 15;
#pragma unroll
    for (int t = 0; t < NT; ++t) {
        const int n = n0 + 16 * t + col;
#pragma unroll
        for (int i = 0; i < 8; ++i) {
            const int m = m0 + (hi << 3) + i;
            const float v = (float)acc[t][i] * deq;
            if (MODE == 0) {
                out8[(size_t)m * ldo + n] = (i8)fq_clampf(v, outScale);
            } else if (MODE == 1) {
                const float q = fq_clampf(v, outScale) * outScale;
                outf[(size_t)m * ldo + n] = softplusf(q + bias[n]);
            } else {
                outf[(size_t)m * ldo + n] = v;
            }
        }
    }
}

__global__ void zero_slots_kernel(unsigned* slots) {
    if (threadIdx.x < 8 && blockIdx.x == 0) slots[threadIdx.x] = 0u;
}

__global__ void absmax_kernel(const float* __restrict__ x, int n,
                              unsigned* __restrict__ slot) {
    __shared__ float red[256];
    float m = 0.0f;
    for (int i = blockIdx.x * blockDim.x + threadIdx.x; i < n;
         i += gridDim.x * blockDim.x)
        m = fmaxf(m, fabsf(x[i]));
    red[threadIdx.x] = m;
    __syncthreads();
    for (int s = 128; s > 0; s >>= 1) {
        if ((int)threadIdx.x < s)
            red[threadIdx.x] = fmaxf(red[threadIdx.x], red[threadIdx.x + s]);
        __syncthreads();
    }
    if (threadIdx.x == 0) atomicMax(slot, __float_as_uint(red[0]));
}

__global__ void quant_w_i8_kernel(const float* __restrict__ w,
                                  const unsigned* __restrict__ amax,
                                  i8* __restrict__ out, int n) {
    const int i = blockIdx.x * blockDim.x + threadIdx.x;
    if (i >= n) return;
    const float s = __uint_as_float(*amax) * (1.0f / 127.0f);
    out[i] = (i8)fq_clampf(w[i], s);
}

__global__ void quant_w_f32_kernel(const float* __restrict__ w,
                                   const unsigned* __restrict__ amax,
                                   float* __restrict__ out, int n) {
    const int i = blockIdx.x * blockDim.x + threadIdx.x;
    if (i >= n) return;
    const float s = __uint_as_float(*amax) * (1.0f / 127.0f);
    out[i] = fq_clampf(w[i], s) * s;
}

__global__ void quant_act_kernel(const float* __restrict__ x, float s,
                                 i8* __restrict__ out, int n) {
    const int i = blockIdx.x * blockDim.x + threadIdx.x;
    if (i >= n) return;
    out[i] = (i8)fq_clampf(x[i], s);
}

__global__ void conv_silu_kernel(const i8* __restrict__ xz_q,
                                 const float* __restrict__ wq,
                                 const float* __restrict__ cb,
                                 i8* __restrict__ xconv_q) {
    const int idx = blockIdx.x * blockDim.x + threadIdx.x;
    if (idx >= BATCH * SEQLEN * D_INNER) return;
    const int d = idx % D_INNER;
    const int t = (idx / D_INNER) % SEQLEN;
    const int b = idx / (D_INNER * SEQLEN);

    float acc = cb[d];
#pragma unroll
    for (int k = 0; k < D_CONV; ++k) {
        const int tt = t - (D_CONV - 1) + k;
        if (tt >= 0) {
            const float xv =
                (float)xz_q[((size_t)(b * SEQLEN + tt)) * (2 * D_INNER) + d] * S_XZ;
            acc += xv * wq[d * D_CONV + k];
        }
    }
    const float v = siluf(acc);
    xconv_q[(size_t)(b * SEQLEN + t) * D_INNER + d] = (i8)fq_clampf(v, S_CONV_OUT);
}

__global__ void scan_kernel(const i8* __restrict__ xconv_q,
                            const float* __restrict__ dt,
                            const i8* __restrict__ xdbl_q,
                            const i8* __restrict__ xz_q,
                            const float* __restrict__ A_log,
                            const float* __restrict__ Dvec,
                            i8* __restrict__ y_q) {
    const int idx = blockIdx.x * blockDim.x + threadIdx.x;
    if (idx >= BATCH * D_INNER * D_STATE) return;
    const int n = idx & (D_STATE - 1);
    const int g = idx >> 4;
    const int d = g % D_INNER;
    const int b = g / D_INNER;

    const float An = -__expf(A_log[d * D_STATE + n]);
    const float Dd = Dvec[d];
    float h = 0.0f;

    for (int t = 0; t < SEQLEN; ++t) {
        const size_t tok = (size_t)(b * SEQLEN + t);
        const float u   = (float)xconv_q[tok * D_INNER + d] * S_CONV_OUT;
        const float dtv = dt[tok * D_INNER + d];
        const i8* xd = xdbl_q + tok * XDIM;
        const float Bv = (float)xd[DT_RANK + n] * S_XDBL;
        const float Cv = (float)xd[DT_RANK + D_STATE + n] * S_XDBL;
        h = __expf(dtv * An) * h + (dtv * u) * Bv;
        float y = h * Cv;
        y += __shfl_xor(y, 1, 32);
        y += __shfl_xor(y, 2, 32);
        y += __shfl_xor(y, 4, 32);
        y += __shfl_xor(y, 8, 32);
        if (n == 0) {
            y += u * Dd;
            const float z = (float)xz_q[tok * (2 * D_INNER) + D_INNER + d] * S_XZ;
            y_q[tok * D_INNER + d] = (i8)fq_clampf(y * siluf(z), S_Y);
        }
    }
}

__global__ void fwht2048_kernel(const i8* __restrict__ in8, float inScale,
                                const float* __restrict__ inf,
                                i8* __restrict__ out8, float outScale,
                                float* __restrict__ outf) {
    __shared__ float buf[2048];
    const size_t base = (size_t)blockIdx.x * 2048;
    for (int i = threadIdx.x; i < 2048; i += blockDim.x)
        buf[i] = in8 ? (float)in8[base + i] * inScale : inf[base + i];
    __syncthreads();
    for (int h = 1; h < 2048; h <<= 1) {
        for (int p = threadIdx.x; p < 1024; p += blockDim.x) {
            const int i = ((p / h) * (h << 1)) + (p % h);
            const int j = i + h;
            const float a = buf[i], b = buf[j];
            buf[i] = a + b;
            buf[j] = a - b;
        }
        __syncthreads();
    }
    const float nrm = 0.022097086912079608f;
    for (int i = threadIdx.x; i < 2048; i += blockDim.x) {
        const float v = buf[i] * nrm;
        if (out8) out8[base + i] = (i8)fq_clampf(v, outScale);
        else      outf[base + i] = v;
    }
}

extern "C" void kernel_launch(void* const* d_in, const int* in_sizes, int n_in,
                              void* d_out, int out_size, void* d_ws, size_t ws_size,
                              hipStream_t stream) {
    (void)in_sizes; (void)n_in; (void)out_size; (void)ws_size;

    const float* hs      = (const float*)d_in[0];
    const float* W_in    = (const float*)d_in[1];
    const float* conv_w  = (const float*)d_in[2];
    const float* conv_b  = (const float*)d_in[3];
    const float* W_x     = (const float*)d_in[4];
    const float* W_dt    = (const float*)d_in[5];
    const float* dt_bias = (const float*)d_in[6];
    const float* A_log   = (const float*)d_in[7];
    const float* Dvec    = (const float*)d_in[8];
    const float* W_out   = (const float*)d_in[9];
    float* out = (float*)d_out;

    char* ws = (char*)d_ws;
    size_t off = 0;
    auto carve = [&](size_t bytes) -> void* {
        void* p = ws + off;
        off += (bytes + 255) & ~(size_t)255;
        return p;
    };
    unsigned* slots  = (unsigned*)carve(8 * sizeof(unsigned));
    i8*    hs_q      = (i8*)   carve((size_t)NTOK * D_MODEL);
    i8*    Winq      = (i8*)   carve((size_t)2 * D_INNER * D_MODEL);
    i8*    xz_q      = (i8*)   carve((size_t)NTOK * 2 * D_INNER);
    float* convwq    = (float*)carve((size_t)D_INNER * D_CONV * 4);
    i8*    xconv_q   = (i8*)   carve((size_t)NTOK * D_INNER);
    i8*    Wxq       = (i8*)   carve((size_t)XDIM * D_INNER);
    i8*    xdbl_q    = (i8*)   carve((size_t)NTOK * XDIM);
    i8*    Wdtq      = (i8*)   carve((size_t)D_INNER * DT_RANK);
    float* dt_f      = (float*)carve((size_t)NTOK * D_INNER * 4);
    i8*    y_q       = (i8*)   carve((size_t)NTOK * D_INNER);
    float* Wh_f      = (float*)carve((size_t)D_MODEL * D_INNER * 4);
    i8*    Whq       = (i8*)   carve((size_t)D_MODEL * D_INNER);
    i8*    yh_q      = (i8*)   carve((size_t)NTOK * D_INNER);

    auto cdiv = [](int a, int b) { return (a + b - 1) / b; };
    auto rblocks = [&](int n) { int b = cdiv(n, 256); return b > 2048 ? 2048 : b; };

    zero_slots_kernel<<<1, 32, 0, stream>>>(slots);

    absmax_kernel<<<rblocks(2 * D_INNER * D_MODEL), 256, 0, stream>>>(W_in,   2 * D_INNER * D_MODEL, slots + 0);
    absmax_kernel<<<rblocks(XDIM * D_INNER),        256, 0, stream>>>(W_x,    XDIM * D_INNER,        slots + 1);
    absmax_kernel<<<rblocks(D_INNER * DT_RANK),     256, 0, stream>>>(W_dt,   D_INNER * DT_RANK,     slots + 2);
    absmax_kernel<<<rblocks(D_INNER * D_CONV),      256, 0, stream>>>(conv_w, D_INNER * D_CONV,      slots + 3);

    quant_act_kernel<<<cdiv(NTOK * D_MODEL, 256), 256, 0, stream>>>(hs, S_IN, hs_q, NTOK * D_MODEL);
    quant_w_i8_kernel<<<cdiv(2 * D_INNER * D_MODEL, 256), 256, 0, stream>>>(W_in, slots + 0, Winq, 2 * D_INNER * D_MODEL);
    quant_w_i8_kernel<<<cdiv(XDIM * D_INNER, 256),        256, 0, stream>>>(W_x,  slots + 1, Wxq,  XDIM * D_INNER);
    quant_w_i8_kernel<<<cdiv(D_INNER * DT_RANK, 256),     256, 0, stream>>>(W_dt, slots + 2, Wdtq, D_INNER * DT_RANK);
    quant_w_f32_kernel<<<cdiv(D_INNER * D_CONV, 256),     256, 0, stream>>>(conv_w, slots + 3, convwq, D_INNER * D_CONV);

    {
        const int groups = (NTOK / 16) * (2 * D_INNER / 64);
        gemm_i8_wmma<0, 4><<<cdiv(groups, 8), 256, 0, stream>>>(
            hs_q, D_MODEL, Winq, D_MODEL, NTOK, 2 * D_INNER, D_MODEL,
            S_IN, slots + 0, S_XZ, nullptr, xz_q, nullptr, 2 * D_INNER);
    }

    conv_silu_kernel<<<cdiv(NTOK * D_INNER, 256), 256, 0, stream>>>(xz_q, convwq, conv_b, xconv_q);

    {
        const int groups = (NTOK / 16) * (XDIM / 32);
        gemm_i8_wmma<0, 2><<<cdiv(groups, 8), 256, 0, stream>>>(
            xconv_q, D_INNER, Wxq, D_INNER, NTOK, XDIM, D_INNER,
            S_CONV_OUT, slots + 1, S_XDBL, nullptr, xdbl_q, nullptr, XDIM);
    }

    {
        const int groups = (NTOK / 16) * (D_INNER / 64);
        gemm_i8_wmma<1, 4><<<cdiv(groups, 8), 256, 0, stream>>>(
            xdbl_q, XDIM, Wdtq, DT_RANK, NTOK, D_INNER, DT_RANK,
            S_XDBL, slots + 2, S_DT, dt_bias, nullptr, dt_f, D_INNER);
    }

    scan_kernel<<<cdiv(BATCH * D_INNER * D_STATE, 256), 256, 0, stream>>>(
        xconv_q, dt_f, xdbl_q, xz_q, A_log, Dvec, y_q);

    fwht2048_kernel<<<D_MODEL, 512, 0, stream>>>(nullptr, 0.0f, W_out, nullptr, 0.0f, Wh_f);
    absmax_kernel<<<rblocks(D_MODEL * D_INNER), 256, 0, stream>>>(Wh_f, D_MODEL * D_INNER, slots + 4);
    quant_w_i8_kernel<<<cdiv(D_MODEL * D_INNER, 256), 256, 0, stream>>>(Wh_f, slots + 4, Whq, D_MODEL * D_INNER);

    fwht2048_kernel<<<NTOK, 512, 0, stream>>>(y_q, S_Y, nullptr, yh_q, S_Y, nullptr);

    {
        const int groups = (NTOK / 16) * (D_MODEL / 64);
        gemm_i8_wmma<2, 4><<<cdiv(groups, 8), 256, 0, stream>>>(
            yh_q, D_INNER, Whq, D_INNER, NTOK, D_MODEL, D_INNER,
            S_Y, slots + 4, 0.0f, nullptr, nullptr, out, D_MODEL);
    }
}
